// SimpleHMT_13778255085652
// MI455X (gfx1250) — matrix operand order repaired and checked by reading, not compiled, not run on hardware
//
#include <hip/hip_runtime.h>
#include <hip/hip_bf16.h>

// ---------------- problem constants ----------------
#define BB 8
#define LL 2048
#define DD 1024
#define KS 32      // sensory length
#define JJ 512

#define TP0 2176   // padded T for pass0 (2050 -> 17*128)
#define TP1 2176   // padded T for pass1 (2082 -> 17*128)
#define TPS 640    // padded T for summarize (514 -> 5*128)
#define TPMAX 2176

#define VT0 2050
#define VT1 2082
#define VTS 514

#define SCALE (1.0f/32.0f)   // 1/sqrt(1024)

typedef __bf16 bf16;
typedef __attribute__((ext_vector_type(16))) __bf16 v16bf;
typedef __attribute__((ext_vector_type(8)))  __bf16 v8bf;
typedef __attribute__((ext_vector_type(8)))  float  v8f;

// explicit global (addrspace 1) pointer types: force global_load/global_store
// codegen (round 2 showed generic pointers degrading to flat_load_b128, which
// also ties up DScnt and forces s_wait_loadcnt_dscnt / s_wait_xcnt stalls)
typedef __attribute__((address_space(1))) const __bf16 gc_bf16;
typedef __attribute__((address_space(1))) __bf16       g_bf16;
typedef __attribute__((address_space(1))) const v8bf   gc_v8bf;
typedef __attribute__((address_space(1))) const float  gc_f32;
typedef __attribute__((address_space(1))) float        g_f32;

// ---------------- generic batched WMMA GEMM ----------------
// C[M,N] = alpha * A[M,K] x Bt[N,K]^T + bias[N]
// 4 waves (128 threads) per block; each wave computes a 64x64 tile
// (4x4 WMMA 16x16x32 tiles, 128 f32 accumulator VGPRs per lane).
// Block tile = 128x128.
// Addressing: uniform base pointer advances along K (SGPR adds); per-lane
// fragment offsets are loop-invariant 32-bit VGPRs -> global_load saddr form.
// MODE 0: store bf16 row-major (ldc)
// MODE 2: store f32 rows in window [row_lo,row_hi) to C (rebased), plus
//         optional single row aux_row -> aux (batch stride auxBatch)
template<int MODE>
__global__ __launch_bounds__(128)
void k_gemm(const bf16* __restrict__ A, long long batchA, int lda,
            const bf16* __restrict__ Bt, long long batchB, int ldb,
            void* __restrict__ Cv, long long batchC, int ldc,
            const float* __restrict__ bias, float alpha, int Kdim,
            int row_lo, int row_hi,
            float* __restrict__ aux, int aux_row, long long auxBatch)
{
  const int lane = threadIdx.x & 31;
  const int wave = threadIdx.x >> 5;   // 0..3
  const int wm = wave & 1;             // 2 waves along M
  const int wn = wave >> 1;            // 2 waves along N
  const int b  = blockIdx.z;
  const int M0 = blockIdx.y * 128 + wm * 64;
  const int N0 = blockIdx.x * 128 + wn * 64;

  gc_bf16* Ag = (gc_bf16*)(unsigned long long)(A  + (size_t)b * batchA);
  gc_bf16* Bg = (gc_bf16*)(unsigned long long)(Bt + (size_t)b * batchB);

  const int l15 = lane & 15;
  // A frag (ISA 7.12.2): lane<16 -> row=lane, K{0..7,16..23}; lane>=16 -> row=lane-16, K{8..15,24..31}
  int aoff[4];
  // B frag from Bt (row-major NxK): lane=col; lane<16 -> K{0..7,16..23}; lane>=16 -> K{8..15,24..31}
  int boff[4];
  #pragma unroll
  for (int i = 0; i < 4; ++i)
    aoff[i] = (M0 + 16 * i + l15) * lda + ((lane & 16) ? 8 : 0);
  #pragma unroll
  for (int j = 0; j < 4; ++j)
    boff[j] = (N0 + 16 * j + l15) * ldb + ((lane & 16) ? 8 : 0);

  v8f acc[4][4] = {};

  gc_bf16* Ak = Ag;   // uniform: advances by 32 elements per K-step (SALU)
  gc_bf16* Bk = Bg;
  for (int k0 = 0; k0 < Kdim; k0 += 32) {
    if (k0 + 64 <= Kdim) {  // prefetch next K-step (gfx1250 prefetch_b8)
      __builtin_prefetch((const void*)(unsigned long long)(Ak + aoff[0] + 32), 0, 1);
      __builtin_prefetch((const void*)(unsigned long long)(Bk + boff[0] + 32), 0, 1);
    }
    v16bf a[4], bf[4];
    #pragma unroll
    for (int i = 0; i < 4; ++i) {
      union { v16bf v; v8bf h[2]; } u;
      u.h[0] = *(gc_v8bf*)(Ak + aoff[i]);
      u.h[1] = *(gc_v8bf*)(Ak + aoff[i] + 16);
      a[i] = u.v;
    }
    #pragma unroll
    for (int j = 0; j < 4; ++j) {
      union { v16bf v; v8bf h[2]; } u;
      u.h[0] = *(gc_v8bf*)(Bk + boff[j]);
      u.h[1] = *(gc_v8bf*)(Bk + boff[j] + 16);
      bf[j] = u.v;
    }
    #pragma unroll
    for (int i = 0; i < 4; ++i)
      #pragma unroll
      for (int j = 0; j < 4; ++j)
        acc[i][j] = __builtin_amdgcn_wmma_f32_16x16x32_bf16(false, a[i], false, bf[j],
                                                            (short)0, acc[i][j], false, false);
    Ak += 32;
    Bk += 32;
  }

  // C layout: lane = col (lane&15); VGPR r -> row r (lanes 0-15) / row r+8 (lanes 16-31)
  const int cn   = lane & 15;
  const int rofs = (lane & 16) ? 8 : 0;
  gc_f32* biasg = bias ? (gc_f32*)(unsigned long long)bias : (gc_f32*)0;
  g_bf16* C0 = (g_bf16*)(unsigned long long)((bf16*)Cv + (size_t)b * batchC);
  g_f32*  C2 = (g_f32*)(unsigned long long)((float*)Cv + (size_t)b * batchC);
  g_f32*  auxg = aux ? (g_f32*)(unsigned long long)(aux + (size_t)b * auxBatch) : (g_f32*)0;
  #pragma unroll
  for (int i = 0; i < 4; ++i) {
    #pragma unroll
    for (int j = 0; j < 4; ++j) {
      const int n = N0 + 16 * j + cn;
      const float bv = biasg ? (float)biasg[n] : 0.0f;
      #pragma unroll
      for (int r = 0; r < 8; ++r) {
        const int m = M0 + 16 * i + rofs + r;
        const float v = acc[i][j][r] * alpha + bv;
        if (MODE == 0) {
          C0[(size_t)m * ldc + n] = (bf16)v;
        } else {
          if (m >= row_lo && m < row_hi)
            C2[(size_t)(m - row_lo) * ldc + n] = v;
          if (auxg && m == aux_row)
            auxg[n] = v;
        }
      }
    }
  }
}

// ---------------- bf16 batched matrix transpose (LDS tiled, coalesced) ------
// src: [rows x cols] row-major  ->  dst: [cols x rows] row-major, per batch z
__global__ __launch_bounds__(256)
void k_transpose(const bf16* __restrict__ src, bf16* __restrict__ dst,
                 int rows, int cols) {
  __shared__ bf16 tile[32][33];   // +1 pad to dodge bank conflicts
  const int b = blockIdx.z;
  gc_bf16* s = (gc_bf16*)(unsigned long long)(src + (size_t)b * rows * cols);
  g_bf16*  d = (g_bf16*)(unsigned long long)(dst + (size_t)b * rows * cols);
  const int r0 = blockIdx.y * 32;
  const int c0 = blockIdx.x * 32;
  const int tx = threadIdx.x & 31;
  const int ty = threadIdx.x >> 5;   // 0..7
  #pragma unroll
  for (int i = ty; i < 32; i += 8)
    tile[i][tx] = s[(size_t)(r0 + i) * cols + (c0 + tx)];
  __syncthreads();
  #pragma unroll
  for (int i = ty; i < 32; i += 8)
    d[(size_t)(c0 + i) * rows + (r0 + tx)] = tile[tx][i];
}

// ---------------- weight transpose + bf16 convert ----------------
__global__ __launch_bounds__(256)
void k_wtrans(const float* __restrict__ W, bf16* __restrict__ WT) {
  size_t idx = (size_t)blockIdx.x * 256 + threadIdx.x;   // idx = n*D + k
  if (idx >= (size_t)DD * DD) return;
  int k = (int)(idx % DD);
  int n = (int)(idx / DD);
  ((g_bf16*)(unsigned long long)WT)[idx] =
      (bf16)((gc_f32*)(unsigned long long)W)[(size_t)k * DD + n];
}

// ---------------- input builders (write full padded buffers) ----------------
__global__ __launch_bounds__(256)
void k_build_pass0(const float* __restrict__ seg0, bf16* __restrict__ X) {
  size_t idx = (size_t)blockIdx.x * 256 + threadIdx.x;   // B*TP0*D
  if (idx >= (size_t)BB * TP0 * DD) return;
  int d = (int)(idx % DD);
  size_t td = idx / DD;
  int t = (int)(td % TP0);
  int b = (int)(td / TP0);
  float v = 0.0f;
  if (t >= 1 && t <= LL) v = seg0[((size_t)b * LL + (t - 1)) * DD + d];
  X[idx] = (bf16)v;
}

__global__ __launch_bounds__(256)
void k_build_sum(const float* __restrict__ seg1, const float* __restrict__ prompt,
                 bf16* __restrict__ X) {
  size_t idx = (size_t)blockIdx.x * 256 + threadIdx.x;   // B*TPS*D
  if (idx >= (size_t)BB * TPS * DD) return;
  int d = (int)(idx % DD);
  size_t td = idx / DD;
  int t = (int)(td % TPS);
  int b = (int)(td / TPS);
  float v = 0.0f;
  if (t == 0 || t == JJ + 1)      v = prompt[d];
  else if (t >= 1 && t <= JJ)     v = seg1[((size_t)b * LL + (t - 1)) * DD + d];
  X[idx] = (bf16)v;
}

__global__ __launch_bounds__(256)
void k_build_pass1(const float* __restrict__ seg0, const float* __restrict__ seg1,
                   const float* __restrict__ P1, bf16* __restrict__ X) {
  size_t idx = (size_t)blockIdx.x * 256 + threadIdx.x;   // B*TP1*D
  if (idx >= (size_t)BB * TP1 * DD) return;
  int d = (int)(idx % DD);
  size_t td = idx / DD;
  int t = (int)(td % TP1);
  int b = (int)(td / TP1);
  float v = 0.0f;
  if (t < KS)                           v = seg0[((size_t)b * LL + (LL - KS + t)) * DD + d];
  else if (t == KS)                     v = P1[(size_t)b * DD + d];
  else if (t >= KS + 1 && t <= KS + LL) v = seg1[((size_t)b * LL + (t - KS - 1)) * DD + d];
  else if (t == KS + LL + 1)            v = P1[(size_t)b * DD + d];
  X[idx] = (bf16)v;
}

// ---------------- row softmax (in place, bf16), masks padded key columns ----
__global__ __launch_bounds__(256)
void k_softmax(bf16* __restrict__ S, int Tp, int validT) {
  const size_t row = blockIdx.x;              // over B*Tp rows (contiguous)
  g_bf16* p = (g_bf16*)(unsigned long long)(S + row * (size_t)Tp);
  __shared__ float red[256];
  const int tid = threadIdx.x;

  float mx = -3.0e38f;
  for (int j = tid; j < validT; j += 256) mx = fmaxf(mx, (float)p[j]);
  red[tid] = mx; __syncthreads();
  for (int s = 128; s > 0; s >>= 1) {
    if (tid < s) red[tid] = fmaxf(red[tid], red[tid + s]);
    __syncthreads();
  }
  mx = red[0]; __syncthreads();

  float sm = 0.0f;
  for (int j = tid; j < validT; j += 256) sm += __expf((float)p[j] - mx);
  red[tid] = sm; __syncthreads();
  for (int s = 128; s > 0; s >>= 1) {
    if (tid < s) red[tid] += red[tid + s];
    __syncthreads();
  }
  const float inv = 1.0f / red[0];
  __syncthreads();

  for (int j = tid; j < Tp; j += 256) {
    float v = (j < validT) ? __expf((float)p[j] - mx) * inv : 0.0f;
    p[j] = (bf16)v;
  }
}

// ---------------- tiny memory-attention path (8x8) ----------------
__global__ __launch_bounds__(256)
void k_vec_gemm(const float* __restrict__ X, const float* __restrict__ W,
                const float* __restrict__ bias, float* __restrict__ Y) {
  int idx = blockIdx.x * 256 + threadIdx.x;   // B*D
  if (idx >= BB * DD) return;
  int n  = idx % DD;
  int bi = idx / DD;
  float s = bias[n];
  const float* x = X + (size_t)bi * DD;
  for (int k = 0; k < DD; ++k) s += x[k] * W[(size_t)k * DD + n];
  Y[idx] = s;
}

__global__ __launch_bounds__(256)
void k_mem_attn(const float* __restrict__ Qn, const float* __restrict__ Kp,
                const float* __restrict__ M1, float* __restrict__ P1) {
  __shared__ float att[BB][BB];
  const int tid = threadIdx.x;
  if (tid < BB * BB) {
    int b = tid >> 3, j = tid & 7;
    float s = 0.0f;
    for (int d = 0; d < DD; ++d) s += Qn[(size_t)b * DD + d] * Kp[(size_t)j * DD + d];
    att[b][j] = s * SCALE;
  }
  __syncthreads();
  if (tid < BB) {
    float mx = -3.0e38f;
    for (int j = 0; j < BB; ++j) mx = fmaxf(mx, att[tid][j]);
    float sm = 0.0f;
    for (int j = 0; j < BB; ++j) { float e = __expf(att[tid][j] - mx); att[tid][j] = e; sm += e; }
    float inv = 1.0f / sm;
    for (int j = 0; j < BB; ++j) att[tid][j] *= inv;
  }
  __syncthreads();
  for (int idx = tid; idx < BB * DD; idx += 256) {
    int b = idx / DD, d = idx % DD;
    float s = 0.0f;
    for (int j = 0; j < BB; ++j) s += att[b][j] * M1[(size_t)j * DD + d];
    P1[idx] = s;
  }
}

// ---------------- host orchestration ----------------
extern "C" void kernel_launch(void* const* d_in, const int* in_sizes, int n_in,
                              void* d_out, int out_size, void* d_ws, size_t ws_size,
                              hipStream_t stream) {
  const float* seg0   = (const float*)d_in[0];
  const float* seg1   = (const float*)d_in[1];
  const float* prompt = (const float*)d_in[2];
  const float* Wq_mem = (const float*)d_in[3];
  const float* bq_mem = (const float*)d_in[4];
  const float* Wk_mem = (const float*)d_in[5];
  const float* bk_mem = (const float*)d_in[6];
  const float* Wq     = (const float*)d_in[7];
  const float* bq     = (const float*)d_in[8];
  const float* Wk     = (const float*)d_in[9];
  const float* bk     = (const float*)d_in[10];
  const float* Wv     = (const float*)d_in[11];
  const float* bv     = (const float*)d_in[12];
  const float* Wo     = (const float*)d_in[13];
  const float* bo     = (const float*)d_in[14];
  float* outF = (float*)d_out;

  char* ws = (char*)d_ws;
  size_t off = 0;
  auto alloc = [&](size_t bytes) -> char* {
    char* p = ws + off;
    off += (bytes + 255) & ~(size_t)255;
    return p;
  };

  bf16* WqT = (bf16*)alloc((size_t)DD * DD * 2);
  bf16* WkT = (bf16*)alloc((size_t)DD * DD * 2);
  bf16* WvT = (bf16*)alloc((size_t)DD * DD * 2);
  bf16* WoT = (bf16*)alloc((size_t)DD * DD * 2);
  bf16* Xb  = (bf16*)alloc((size_t)BB * TPMAX * DD * 2);
  bf16* Qb  = (bf16*)alloc((size_t)BB * TPMAX * DD * 2);
  bf16* Kb  = (bf16*)alloc((size_t)BB * TPMAX * DD * 2);
  bf16* Vt  = (bf16*)alloc((size_t)BB * DD * TPMAX * 2);
  bf16* Sb  = (bf16*)alloc((size_t)BB * TPMAX * TPMAX * 2);
  bf16* Ob  = (bf16*)alloc((size_t)BB * TPMAX * DD * 2);
  float* M1 = (float*)alloc((size_t)BB * DD * 4);
  float* S1 = (float*)alloc((size_t)BB * DD * 4);
  float* Qn = (float*)alloc((size_t)BB * DD * 4);
  float* Kp = (float*)alloc((size_t)BB * DD * 4);
  float* P1 = (float*)alloc((size_t)BB * DD * 4);

  // weight transposes (bf16)
  {
    int g = (DD * DD) / 256;
    k_wtrans<<<g, 256, 0, stream>>>(Wq, WqT);
    k_wtrans<<<g, 256, 0, stream>>>(Wk, WkT);
    k_wtrans<<<g, 256, 0, stream>>>(Wv, WvT);
    k_wtrans<<<g, 256, 0, stream>>>(Wo, WoT);
  }

  // full backbone pass over padded X (Tp rows/batch, validT real rows)
  auto backbone = [&](const bf16* X, int Tp, int validT,
                      float* ydst, long long ybatch, int row_lo, int row_hi,
                      float* aux, int aux_row) {
    const long long XA  = (long long)Tp * DD;   // batch stride X/Q/K/V/O
    const long long VB  = (long long)DD * Tp;   // batch stride Vt
    const long long SBs = (long long)Tp * Tp;   // batch stride scores
    dim3 gProj(DD / 128, Tp / 128, BB);
    dim3 gScore(Tp / 128, Tp / 128, BB);
    dim3 gT(DD / 32, Tp / 32, BB);
    // Q, K, V (bf16 row-major); V uses Ob as scratch then transposed -> Vt
    k_gemm<0><<<gProj, 128, 0, stream>>>(X, XA, DD, WqT, 0, DD, Qb, XA, DD, bq, 1.0f, DD, 0, 0, nullptr, 0, 0);
    k_gemm<0><<<gProj, 128, 0, stream>>>(X, XA, DD, WkT, 0, DD, Kb, XA, DD, bk, 1.0f, DD, 0, 0, nullptr, 0, 0);
    k_gemm<0><<<gProj, 128, 0, stream>>>(X, XA, DD, WvT, 0, DD, Ob, XA, DD, bv, 1.0f, DD, 0, 0, nullptr, 0, 0);
    k_transpose<<<gT, 256, 0, stream>>>(Ob, Vt, Tp, DD);
    // S = scale * Q K^T   (Bt = K row-major: N=keys, K=D)
    k_gemm<0><<<gScore, 128, 0, stream>>>(Qb, XA, DD, Kb, XA, DD, Sb, SBs, Tp, nullptr, SCALE, DD, 0, 0, nullptr, 0, 0);
    // softmax rows (mask padded key cols)
    k_softmax<<<BB * Tp, 256, 0, stream>>>(Sb, Tp, validT);
    // O = P V   (Bt = Vt: N=D, K=keys)
    k_gemm<0><<<gProj, 128, 0, stream>>>(Sb, SBs, Tp, Vt, VB, Tp, Ob, XA, DD, nullptr, 1.0f, Tp, 0, 0, nullptr, 0, 0);
    // Y = O Wo + bo, f32 store of window rows (+optional aux row)
    k_gemm<2><<<gProj, 128, 0, stream>>>(Ob, XA, DD, WoT, 0, DD, ydst, ybatch, DD, bo, 1.0f, DD, row_lo, row_hi, aux, aux_row, (long long)DD);
  };

  // ---- pass 0: input_h0 = [0, seg0, 0]; out0 = rows 33..2048, M1 = row 2049
  k_build_pass0<<<(int)(((size_t)BB * TP0 * DD) / 256), 256, 0, stream>>>(seg0, Xb);
  backbone(Xb, TP0, VT0, outF, (long long)2016 * DD, KS + 1, LL + 1, M1, LL + 1);

  // ---- summarize(seg1): inp = [T, seg1[:512], T]; S1 = row 512
  k_build_sum<<<(int)(((size_t)BB * TPS * DD) / 256), 256, 0, stream>>>(seg1, prompt, Xb);
  backbone(Xb, TPS, VTS, S1, (long long)DD, JJ, JJ + 1, nullptr, -1);

  // ---- memory attention (tiny): Qn=S1 Wq_mem+b; Kp=M1 Wk_mem+b; P1=softmax(Qn Kp^T/32) M1
  k_vec_gemm<<<(BB * DD) / 256, 256, 0, stream>>>(S1, Wq_mem, bq_mem, Qn);
  k_vec_gemm<<<(BB * DD) / 256, 256, 0, stream>>>(M1, Wk_mem, bk_mem, Kp);
  k_mem_attn<<<1, 256, 0, stream>>>(Qn, Kp, M1, P1);

  // ---- pass 1: input_h1 = [seg0[-32:], P1, seg1, P1]; out1 = rows 33..2080
  k_build_pass1<<<(int)(((size_t)BB * TP1 * DD) / 256), 256, 0, stream>>>(seg0, seg1, P1, Xb);
  backbone(Xb, TP1, VT1, outF + (size_t)BB * 2016 * DD, (long long)2048 * DD, KS + 1, KS + LL + 1, nullptr, -1);
}
